// SemanticsModulatedAttention_68831145885962
// MI455X (gfx1250) — matrix operand order repaired and checked by reading, not compiled, not run on hardware
//
#include <hip/hip_runtime.h>
#include <hip/hip_bf16.h>

// ---------------- problem constants ----------------
constexpr int B_    = 32;
constexpr int T_    = 512;
constexpr int D_    = 1024;
constexpr int NT_   = 77;
constexpr int TL_   = 768;
constexpr int H_    = 8;
constexpr int TE_   = 2048;
constexpr int DH_   = 128;          // D/H
constexpr int TWOD_ = 2048;         // 2*D
constexpr int NTOT_ = NT_ + T_;     // 589
constexpr int NPAD_ = 608;          // 19*32, K-padded for WMMA
constexpr int NTILE_ = NPAD_ / 32;  // 19
constexpr int BT_   = B_ * T_;      // 16384
constexpr int BNT_  = B_ * NT_;     // 2464
constexpr int BH_   = B_ * H_;      // 256
constexpr float NEG_ = -1000000.0f;

typedef __bf16 bf16_t;
typedef __attribute__((ext_vector_type(16))) __bf16 v16bf;
typedef __attribute__((ext_vector_type(8)))  __bf16 v8bf;
typedef __attribute__((ext_vector_type(8)))  float  v8f;

// ---------------- CDNA5 async memory->LDS copy (ASYNCcnt-tracked DMA) ----------------
// Per-lane: 16B from global VA to LDS. LDS address = low 32 bits of the generic
// shared-memory VA (ISA 10.2: LDS aperture VAs carry the LDS offset in addr[31:0]).
__device__ __forceinline__ void async_copy_b128(const bf16_t* g, bf16_t* l)
{
    asm volatile("global_load_async_to_lds_b128 %0, %1, off"
                 :: "v"((unsigned)(size_t)l), "v"(g)
                 : "memory");
}
__device__ __forceinline__ void wait_async0()
{
    asm volatile("s_wait_asynccnt 0" ::: "memory");
}

// ---------------- block reductions (blockDim == 256) ----------------
__device__ __forceinline__ float block_sum(float v, float* red) {
    int tid = threadIdx.x;
    red[tid] = v; __syncthreads();
    for (int off = 128; off > 0; off >>= 1) {
        if (tid < off) red[tid] += red[tid + off];
        __syncthreads();
    }
    float r = red[0]; __syncthreads();
    return r;
}
__device__ __forceinline__ float block_max(float v, float* red) {
    int tid = threadIdx.x;
    red[tid] = v; __syncthreads();
    for (int off = 128; off > 0; off >>= 1) {
        if (tid < off) red[tid] = fmaxf(red[tid], red[tid + off]);
        __syncthreads();
    }
    float r = red[0]; __syncthreads();
    return r;
}

// ---------------- generic WMMA bf16 GEMM (B pre-transposed) ----------------
// C[z, m, n] = sum_k A[z, m, k] * B[k, n]  with B supplied as BT (N x K row-major).
// Tile 128x128, K-step 32, double-buffered LDS refilled by async-to-LDS DMA,
// 8 waves each owning 32x64 (2x4 WMMA 16x16x32 frags).
constexpr int LDSS = 40;                 // 32 + 8 pad (bf16 elems), 80B rows
constexpr int TILE_ELEMS = 128 * LDSS;

template<bool GUARD, bool HASB, bool HASR>
__device__ __forceinline__ void gemm_core(
    const bf16_t* __restrict__ A, int M, int K,
    const bf16_t* __restrict__ BT,
    float* __restrict__ C, int ldc,
    const float* __restrict__ bias,
    const float* __restrict__ resid, int ldr,
    int tileM, int tileN, bf16_t* As, bf16_t* Bs)
{
    const int tid  = threadIdx.x;
    const int lane = tid & 31;
    const int wave = tid >> 5;
    const int wm   = wave & 3;           // 32-row strip
    const int wn   = wave >> 2;          // 64-col strip
    const int half = lane >> 4;          // lanes 0-15 vs 16-31
    const int l15  = lane & 15;

    // staging map: each thread owns one 32B run (16 bf16) of a 128x32 tile
    const int sRow = tid >> 1;           // 0..127
    const int sCol = (tid & 1) * 16;     // 0 or 16
    const bool aRowOk = !GUARD || (tileM + sRow) < M;

    // guarded tiles: masked lanes never DMA their A rows -> zero them once (both buffers)
    if (GUARD && !aRowOk) {
        uint4 z = make_uint4(0u, 0u, 0u, 0u);
#pragma unroll
        for (int bfi = 0; bfi < 2; ++bfi) {
            *(uint4*)(As + bfi * TILE_ELEMS + sRow * LDSS + sCol)     = z;
            *(uint4*)(As + bfi * TILE_ELEMS + sRow * LDSS + sCol + 8) = z;
        }
    }

    auto issue = [&](int k0, int buf) {
        bf16_t* la = As + buf * TILE_ELEMS + sRow * LDSS + sCol;
        bf16_t* lb = Bs + buf * TILE_ELEMS + sRow * LDSS + sCol;
        if (aRowOk) {
            const bf16_t* ga = A + (long)(tileM + sRow) * K + k0 + sCol;
            async_copy_b128(ga,     la);
            async_copy_b128(ga + 8, la + 8);
        }
        const bf16_t* gb = BT + (long)(tileN + sRow) * K + k0 + sCol;
        async_copy_b128(gb,     lb);
        async_copy_b128(gb + 8, lb + 8);
    };

    v8f zero = {};
    v8f acc[2][4];
#pragma unroll
    for (int i = 0; i < 2; ++i)
#pragma unroll
        for (int j = 0; j < 4; ++j) acc[i][j] = zero;

    issue(0, 0);

    const int steps = K >> 5;
    int cur = 0;
    for (int s = 0; s < steps; ++s) {
        wait_async0();                       // this wave's DMA for LDS[cur] done
        __syncthreads();                     // all waves' DMA published
        if (s + 1 < steps) issue((s + 1) << 5, cur ^ 1);
        if (s + 2 < steps && aRowOk) {
            __builtin_prefetch((const void*)(A + (long)(tileM + sRow) * K + ((s + 2) << 5) + sCol), 0, 1);
            __builtin_prefetch((const void*)(BT + (long)(tileN + sRow) * K + ((s + 2) << 5) + sCol), 0, 1);
        }

        const bf16_t* cA = As + cur * TILE_ELEMS;
        const bf16_t* cB = Bs + cur * TILE_ELEMS;

        // A frags: lanes 0-15 hold row, K {0..7,16..23}; lanes 16-31 K {8..15,24..31}
        v16bf af[2];
#pragma unroll
        for (int i = 0; i < 2; ++i) {
            const bf16_t* p = cA + (wm * 32 + i * 16 + l15) * LDSS;
            v8bf lo = *(const v8bf*)(p + (half ? 8 : 0));
            v8bf hi = *(const v8bf*)(p + (half ? 24 : 16));
            af[i] = __builtin_shufflevector(lo, hi, 0,1,2,3,4,5,6,7,8,9,10,11,12,13,14,15);
        }
        // B frags: lane = column; lanes 0-15 K {0..7,16..23}; lanes 16-31 K {8..15,24..31}
        v16bf bfr[4];
#pragma unroll
        for (int j = 0; j < 4; ++j) {
            const bf16_t* p = cB + (wn * 64 + j * 16 + l15) * LDSS + (half ? 8 : 0);
            v8bf lo = *(const v8bf*)(p);
            v8bf hi = *(const v8bf*)(p + 16);
            bfr[j] = __builtin_shufflevector(lo, hi, 0,1,2,3,4,5,6,7,8,9,10,11,12,13,14,15);
        }
#pragma unroll
        for (int i = 0; i < 2; ++i)
#pragma unroll
            for (int j = 0; j < 4; ++j)
                acc[i][j] = __builtin_amdgcn_wmma_f32_16x16x32_bf16(
                    false, af[i], false, bfr[j], (short)0, acc[i][j], false, false);

        cur ^= 1;
    }

    // store: C/D frag layout: VGPR r, lanes0-15 -> M=r, lanes16-31 -> M=8+r; N=lane&15
#pragma unroll
    for (int i = 0; i < 2; ++i) {
#pragma unroll
        for (int j = 0; j < 4; ++j) {
            int colg = tileN + wn * 64 + j * 16 + l15;
#pragma unroll
            for (int r = 0; r < 8; ++r) {
                int rowg = tileM + wm * 32 + i * 16 + r + half * 8;
                if (!GUARD || rowg < M) {
                    float v = acc[i][j][r];
                    if (HASB) v += bias[colg];
                    if (HASR) v += resid[(long)rowg * ldr + colg];
                    C[(long)rowg * ldc + colg] = v;
                }
            }
        }
    }
}

template<bool HASB, bool HASR>
__global__ __launch_bounds__(256)
void gemm_bf16_wmma(const bf16_t* __restrict__ A, long sAz, int M, int K,
                    const bf16_t* __restrict__ BT, long sBz,
                    float* __restrict__ C, long sCz, int ldc,
                    const float* __restrict__ bias,
                    const float* __restrict__ resid, int ldr)
{
    __shared__ __align__(16) bf16_t As[2 * TILE_ELEMS];
    __shared__ __align__(16) bf16_t Bs[2 * TILE_ELEMS];
    const int z = blockIdx.z;
    const bf16_t* Az = A  + (long)z * sAz;
    const bf16_t* Bz = BT + (long)z * sBz;
    float*        Cz = C  + (long)z * sCz;
    const int tileM = blockIdx.y * 128;
    const int tileN = blockIdx.x * 128;
    if (tileM + 128 <= M)
        gemm_core<false, HASB, HASR>(Az, M, K, Bz, Cz, ldc, bias, resid, ldr, tileM, tileN, As, Bs);
    else
        gemm_core<true, HASB, HASR>(Az, M, K, Bz, Cz, ldc, bias, resid, ldr, tileM, tileN, As, Bs);
}

// ---------------- tiled transpose + f32->bf16: W (K x N) -> WT (N x K) ----------------
__global__ __launch_bounds__(256)
void transpose_to_bf16(const float* __restrict__ W, bf16_t* __restrict__ WT, int K, int N)
{
    __shared__ float tile[32][33];
    const int kt = blockIdx.y * 32, nt = blockIdx.x * 32;
    const int tx = threadIdx.x & 31, ty = threadIdx.x >> 5;   // ty 0..7
#pragma unroll
    for (int i = 0; i < 4; ++i)
        tile[ty + i * 8][tx] = W[(long)(kt + ty + i * 8) * N + nt + tx];
    __syncthreads();
#pragma unroll
    for (int i = 0; i < 4; ++i)
        WT[(long)(nt + ty + i * 8) * K + kt + tx] = (bf16_t)tile[tx][ty + i * 8];
}

// ---------------- LayerNorm (f32 in) -> bf16 out ----------------
__global__ __launch_bounds__(256)
void layernorm_bf16(const float* __restrict__ X, const float* __restrict__ g,
                    const float* __restrict__ b, bf16_t* __restrict__ out, int W)
{
    __shared__ float red[256];
    const long row = blockIdx.x;
    const float* xr = X + row * W;
    float s = 0.f;
    for (int i = threadIdx.x; i < W; i += 256) s += xr[i];
    float mean = block_sum(s, red) / (float)W;
    float s2 = 0.f;
    for (int i = threadIdx.x; i < W; i += 256) { float d = xr[i] - mean; s2 += d * d; }
    float inv = rsqrtf(block_sum(s2, red) / (float)W + 1e-5f);
    for (int i = threadIdx.x; i < W; i += 256) {
        float v = (xr[i] - mean) * inv * g[i] + b[i];
        out[row * W + i] = (bf16_t)v;
    }
}

// ---------------- f32 -> bf16 elementwise ----------------
__global__ void f32_to_bf16(const float* __restrict__ in, bf16_t* __restrict__ out, long n)
{
    long i = (long)blockIdx.x * blockDim.x + threadIdx.x;
    if (i < n) out[i] = (bf16_t)in[i];
}

// ---------------- silu(emb) -> bf16 ----------------
__global__ void silu_bf16(const float* __restrict__ in, bf16_t* __restrict__ out, long n)
{
    long i = (long)blockIdx.x * blockDim.x + threadIdx.x;
    if (i < n) {
        float v = in[i];
        out[i] = (bf16_t)(v / (1.f + __expf(-v)));
    }
}

// ---------------- q softmax over last 128, reorder to (B,H,T,128) bf16 ----------------
__global__ __launch_bounds__(256)
void softmax_q(const float* __restrict__ q, bf16_t* __restrict__ qh)
{
    const long row = (long)blockIdx.x * 8 + (threadIdx.x >> 5);   // (b*T+t)*H + h
    const int lane = threadIdx.x & 31;
    const long bt = row / H_;
    const int  h  = (int)(row % H_);
    const int  b  = (int)(bt / T_);
    const int  t  = (int)(bt % T_);
    const float* p = q + bt * D_ + h * DH_;
    float v[4];
    float mx = -1e30f;
#pragma unroll
    for (int i = 0; i < 4; ++i) { v[i] = p[lane + 32 * i]; mx = fmaxf(mx, v[i]); }
    for (int o = 16; o > 0; o >>= 1) mx = fmaxf(mx, __shfl_xor(mx, o, 32));
    float sum = 0.f;
#pragma unroll
    for (int i = 0; i < 4; ++i) { v[i] = __expf(v[i] - mx); sum += v[i]; }
    for (int o = 16; o > 0; o >>= 1) sum += __shfl_xor(sum, o, 32);
    float inv = 1.f / sum;
    bf16_t* o2 = qh + (((long)(b * H_ + h) * T_) + t) * DH_;
#pragma unroll
    for (int i = 0; i < 4; ++i) o2[lane + 32 * i] = (bf16_t)(v[i] * inv);
}

// ---------------- k column softmax over N=589 (concat kt|km, masked) -> kT bf16 (B,H,128,NPAD) ----------------
__global__ __launch_bounds__(256)
void k_softmax(const float* __restrict__ kt, const float* __restrict__ km,
               const int* __restrict__ mask, const int* __restrict__ cond,
               bf16_t* __restrict__ kT)
{
    __shared__ float red[256];
    const int b = blockIdx.x / D_;
    const int c = blockIdx.x % D_;
    const float tct  = (cond[0] % 10 == 0) ? 0.f : 1.f;
    const float negT = (1.f - tct) * NEG_;
    const int tid = threadIdx.x;

    float vals[3];
    float mx = -1e30f;
#pragma unroll
    for (int it = 0; it < 3; ++it) {
        int n = tid + it * 256;
        float val = -1e30f;
        if (n < NTOT_) {
            if (n < NT_) {
                val = kt[((long)b * NT_ + n) * D_ + c] + negT;
            } else {
                int t = n - NT_;
                float mm = (float)mask[b * T_ + t];
                val = km[((long)b * T_ + t) * D_ + c] + (1.f - mm) * NEG_;
            }
        }
        vals[it] = val;
        mx = fmaxf(mx, val);
    }
    mx = block_max(mx, red);
    float s = 0.f;
#pragma unroll
    for (int it = 0; it < 3; ++it) {
        int n = tid + it * 256;
        float e = (n < NTOT_) ? __expf(vals[it] - mx) : 0.f;
        vals[it] = e; s += e;
    }
    float inv = 1.f / block_sum(s, red);
    const int h = c / DH_, d = c % DH_;
    bf16_t* outp = kT + (((long)(b * H_ + h) * DH_) + d) * NPAD_;
#pragma unroll
    for (int it = 0; it < 3; ++it) {
        int n = tid + it * 256;
        if (n < NPAD_) outp[n] = (bf16_t)((n < NTOT_) ? vals[it] * inv : 0.f);
    }
}

// ---------------- build vT (B,H,128,NPAD) bf16 with tct / mask scaling (LDS tile transpose) ----------------
__global__ __launch_bounds__(256)
void v_buildT(const float* __restrict__ vt, const float* __restrict__ vm,
              const int* __restrict__ mask, const int* __restrict__ cond,
              bf16_t* __restrict__ vT)
{
    __shared__ bf16_t tile[32][DH_ + 8];
    const int blk = blockIdx.x;
    const int ntile = blk % NTILE_;
    const int bh = blk / NTILE_;
    const int h = bh % H_, b = bh / H_;
    const float tct = (cond[0] % 10 == 0) ? 0.f : 1.f;
    const int tid = threadIdx.x;
#pragma unroll
    for (int p = 0; p < 16; ++p) {
        int e  = p * 256 + tid;
        int no = e >> 7;           // 0..31
        int l  = e & 127;
        int n  = ntile * 32 + no;
        float val = 0.f;
        if (n < NT_) {
            val = vt[((long)b * NT_ + n) * D_ + h * DH_ + l] * tct;
        } else if (n < NTOT_) {
            int t = n - NT_;
            val = vm[((long)b * T_ + t) * D_ + h * DH_ + l] * (float)mask[b * T_ + t];
        }
        tile[no][l] = (bf16_t)val;
    }
    __syncthreads();
#pragma unroll
    for (int p = 0; p < 16; ++p) {
        int e  = p * 256 + tid;
        int no = e & 31;
        int d  = e >> 5;           // 0..127
        vT[(((long)(b * H_ + h) * DH_) + d) * NPAD_ + ntile * 32 + no] = tile[no][d];
    }
}

// ---------------- AdaLN-modulate + silu -> hmod bf16 (BT, D) ----------------
__global__ __launch_bounds__(256)
void modulate_silu(const float* __restrict__ y, const float* __restrict__ eo,
                   const float* __restrict__ g, const float* __restrict__ bb,
                   bf16_t* __restrict__ out)
{
    __shared__ float red[256];
    const int bt = blockIdx.x;
    const int b = bt / T_, t = bt % T_;
    float loc[4];
    float s = 0.f;
#pragma unroll
    for (int i = 0; i < 4; ++i) {
        int d2 = threadIdx.x + 256 * i;
        int h = d2 >> 7, l = d2 & 127;
        float v = y[(((long)(b * H_ + h) * T_) + t) * DH_ + l];
        loc[i] = v; s += v;
    }
    float mean = block_sum(s, red) / (float)D_;
    float s2 = 0.f;
#pragma unroll
    for (int i = 0; i < 4; ++i) { float d = loc[i] - mean; s2 += d * d; }
    float inv = rsqrtf(block_sum(s2, red) / (float)D_ + 1e-5f);
#pragma unroll
    for (int i = 0; i < 4; ++i) {
        int d2 = threadIdx.x + 256 * i;
        float v = (loc[i] - mean) * inv * g[d2] + bb[d2];
        float sc = eo[(long)b * TWOD_ + d2];
        float sh = eo[(long)b * TWOD_ + D_ + d2];
        v = v * (1.f + sc) + sh;
        v = v / (1.f + __expf(-v));               // silu
        out[(long)bt * D_ + d2] = (bf16_t)v;
    }
}

// ---------------- host-side orchestration ----------------
static inline size_t alignup(size_t x) { return (x + 255) & ~(size_t)255; }

extern "C" void kernel_launch(void* const* d_in, const int* in_sizes, int n_in,
                              void* d_out, int out_size, void* d_ws, size_t ws_size,
                              hipStream_t stream)
{
    (void)in_sizes; (void)n_in; (void)out_size; (void)ws_size;
    const float* x    = (const float*)d_in[0];
    const float* xf   = (const float*)d_in[1];
    const float* emb  = (const float*)d_in[2];
    const int*   msk  = (const int*)  d_in[3];
    const int*   cond = (const int*)  d_in[4];
    const float* g_n  = (const float*)d_in[5];
    const float* b_n  = (const float*)d_in[6];
    const float* g_tn = (const float*)d_in[7];
    const float* b_tn = (const float*)d_in[8];
    const float* Wq   = (const float*)d_in[9];
    const float* bq   = (const float*)d_in[10];
    const float* Wkt  = (const float*)d_in[11];
    const float* bkt  = (const float*)d_in[12];
    const float* Wvt  = (const float*)d_in[13];
    const float* bvt  = (const float*)d_in[14];
    const float* Wkm  = (const float*)d_in[15];
    const float* bkm  = (const float*)d_in[16];
    const float* Wvm  = (const float*)d_in[17];
    const float* bvm  = (const float*)d_in[18];
    const float* We   = (const float*)d_in[19];
    const float* be   = (const float*)d_in[20];
    const float* g_s  = (const float*)d_in[21];
    const float* b_s  = (const float*)d_in[22];
    const float* Wo   = (const float*)d_in[23];
    const float* bo   = (const float*)d_in[24];
    float* out = (float*)d_out;
    char*  ws  = (char*)d_ws;

    // workspace layout (with live-range reuse)
    size_t off = 0;
    auto take = [&](size_t bytes) { size_t r = off; off += alignup(bytes); return r; };
    bf16_t* xn_bf  = (bf16_t*)(ws + take((size_t)BT_  * D_  * 2));
    bf16_t* xfn_bf = (bf16_t*)(ws + take((size_t)BNT_ * TL_ * 2));
    bf16_t* wbuf   = (bf16_t*)(ws + take((size_t)TE_  * TWOD_ * 2));   // largest WT (We^T)
    bf16_t* qh_bf  = (bf16_t*)(ws + take((size_t)BH_  * T_ * DH_ * 2));
    char*   slabA  = ws + take((size_t)BT_ * D_ * 4);                  // qf, later yf
    char*   slabB  = ws + take((size_t)BT_ * D_ * 4);                  // kmf, later hmod_bf
    char*   slabC  = ws + take((size_t)BT_ * D_ * 4);                  // vmf, later attTf + attT_bf
    float*  ktf    = (float*)(ws + take((size_t)BNT_ * D_ * 4));
    float*  vtf    = (float*)(ws + take((size_t)BNT_ * D_ * 4));
    bf16_t* kT_bf  = (bf16_t*)(ws + take((size_t)BH_ * DH_ * NPAD_ * 2));
    bf16_t* vT_bf  = (bf16_t*)(ws + take((size_t)BH_ * DH_ * NPAD_ * 2));
    bf16_t* se_bf  = (bf16_t*)(ws + take((size_t)B_ * TE_ * 2));
    float*  eof    = (float*)(ws + take((size_t)B_ * TWOD_ * 4));

    float*  qf      = (float*)slabA;
    float*  yf      = (float*)slabA;
    float*  kmf     = (float*)slabB;
    bf16_t* hmod_bf = (bf16_t*)slabB;
    float*  vmf     = (float*)slabC;
    float*  attTf   = (float*)slabC;
    bf16_t* attT_bf = (bf16_t*)(slabC + alignup((size_t)BH_ * DH_ * DH_ * 4));

    auto cvtT = [&](const float* W, bf16_t* WT, int K, int N) {
        transpose_to_bf16<<<dim3(N / 32, K / 32), 256, 0, stream>>>(W, WT, K, N);
    };

    // 1) LayerNorms -> bf16
    layernorm_bf16<<<BT_,  256, 0, stream>>>(x,  g_n,  b_n,  xn_bf,  D_);
    layernorm_bf16<<<BNT_, 256, 0, stream>>>(xf, g_tn, b_tn, xfn_bf, TL_);

    // 2) Q = xn@Wq + bq ; softmax -> qh (B,H,T,128)
    cvtT(Wq, wbuf, D_, D_);
    gemm_bf16_wmma<true, false><<<dim3(D_ / 128, BT_ / 128, 1), 256, 0, stream>>>(
        xn_bf, 0, BT_, D_, wbuf, 0, qf, 0, D_, bq, nullptr, 0);
    softmax_q<<<(BT_ * H_) / 8, 256, 0, stream>>>(qf, qh_bf);

    // 3) Km, Kt -> column softmax -> kT (B,H,128,NPAD)
    cvtT(Wkm, wbuf, D_, D_);
    gemm_bf16_wmma<true, false><<<dim3(D_ / 128, BT_ / 128, 1), 256, 0, stream>>>(
        xn_bf, 0, BT_, D_, wbuf, 0, kmf, 0, D_, bkm, nullptr, 0);
    cvtT(Wkt, wbuf, TL_, D_);
    gemm_bf16_wmma<true, false><<<dim3(D_ / 128, (BNT_ + 127) / 128, 1), 256, 0, stream>>>(
        xfn_bf, 0, BNT_, TL_, wbuf, 0, ktf, 0, D_, bkt, nullptr, 0);
    k_softmax<<<B_ * D_, 256, 0, stream>>>(ktf, kmf, msk, cond, kT_bf);

    // 4) Vm, Vt -> vT (B,H,128,NPAD)
    cvtT(Wvm, wbuf, D_, D_);
    gemm_bf16_wmma<true, false><<<dim3(D_ / 128, BT_ / 128, 1), 256, 0, stream>>>(
        xn_bf, 0, BT_, D_, wbuf, 0, vmf, 0, D_, bvm, nullptr, 0);
    cvtT(Wvt, wbuf, TL_, D_);
    gemm_bf16_wmma<true, false><<<dim3(D_ / 128, (BNT_ + 127) / 128, 1), 256, 0, stream>>>(
        xfn_bf, 0, BNT_, TL_, wbuf, 0, vtf, 0, D_, bvt, nullptr, 0);
    v_buildT<<<B_ * H_ * NTILE_, 256, 0, stream>>>(vtf, vmf, msk, cond, vT_bf);

    // 5) attT[z] = vT (128 x NPAD) @ k  (BT operand = kT, 128 x NPAD) -> attT = att^T
    gemm_bf16_wmma<false, false><<<dim3(1, 1, BH_), 256, 0, stream>>>(
        vT_bf, (long)DH_ * NPAD_, DH_, NPAD_,
        kT_bf, (long)DH_ * NPAD_,
        attTf, (long)DH_ * DH_, DH_, nullptr, nullptr, 0);
    f32_to_bf16<<<(unsigned)(((long)BH_ * DH_ * DH_ + 255) / 256), 256, 0, stream>>>(
        attTf, attT_bf, (long)BH_ * DH_ * DH_);

    // 6) y[z] = qh (T x 128) @ att  (BT operand = attT) -> (B,H,T,128) f32
    gemm_bf16_wmma<false, false><<<dim3(1, T_ / 128, BH_), 256, 0, stream>>>(
        qh_bf, (long)T_ * DH_, T_, DH_,
        attT_bf, (long)DH_ * DH_,
        yf, (long)T_ * DH_, DH_, nullptr, nullptr, 0);

    // 7) eo = silu(emb) @ We + be
    silu_bf16<<<(B_ * TE_ + 255) / 256, 256, 0, stream>>>(emb, se_bf, (long)B_ * TE_);
    cvtT(We, wbuf, TE_, TWOD_);
    gemm_bf16_wmma<true, false><<<dim3(TWOD_ / 128, 1, 1), 256, 0, stream>>>(
        se_bf, 0, B_, TE_, wbuf, 0, eof, 0, TWOD_, be, nullptr, 0);

    // 8) h = silu(ln(y)*(1+scale)+shift) -> bf16
    modulate_silu<<<BT_, 256, 0, stream>>>(yf, eof, g_s, b_s, hmod_bf);

    // 9) out = x + h @ Wo + bo
    cvtT(Wo, wbuf, D_, D_);
    gemm_bf16_wmma<true, true><<<dim3(D_ / 128, BT_ / 128, 1), 256, 0, stream>>>(
        hmod_bf, 0, BT_, D_, wbuf, 0, out, 0, D_, bo, x, D_);
}
